// DCondJastrow_48627619726083
// MI455X (gfx1250) — matrix operand order repaired and checked by reading, not compiled, not run on hardware
//
#include <hip/hip_runtime.h>
#include <hip/hip_bf16.h>

typedef __attribute__((ext_vector_type(16))) _Float16 v16h;
typedef __attribute__((ext_vector_type(8)))  float    v8f;

#define BATCH 16384
#define NPART 16
#define NPAIR 120
#define HID   64
#define DL    5
#define DEMB  16
#define CUSP_GAMMA 1.0f   /* 1/(DIM-1), DIM=2 */

#define LOG2E  1.4426950408889634f
#define LN2    0.6931471805599453f

// Branch-free erf (Abramowitz-Stegun 7.1.26, |abs err| < 1.5e-7), raw TRANS ops.
__device__ __forceinline__ float erf_fast(float u) {
    const float au = fabsf(u);
    const float t  = __builtin_amdgcn_rcpf(fmaf(0.3275911f, au, 1.0f));
    float p = 1.061405429f;
    p = fmaf(p, t, -1.453152027f);
    p = fmaf(p, t,  1.421413741f);
    p = fmaf(p, t, -0.284496736f);
    p = fmaf(p, t,  0.254829592f);
    p = p * t;
    const float e = __builtin_amdgcn_exp2f(-au * au * LOG2E);   // exp(-u^2)
    const float erfabs = fmaf(-p, e, 1.0f);
    return copysignf(erfabs, u);
}

__device__ __forceinline__ float gelu_exact(float x) {
    return 0.5f * x * (1.0f + erf_fast(x * 0.70710678118654752f));
}

__device__ __forceinline__ v8f wmma_f32_f16(v16h a, v16h b, v8f c) {
    return __builtin_amdgcn_wmma_f32_16x16x32_f16(false, a, false, b, (short)0, c, false, false);
}

__device__ __forceinline__ v16h load_btile(const _Float16* packed, int tile, int lane) {
    // tile is 32 lanes x 16 halves, lane-contiguous: one 32B vector load per lane
    return ((const v16h*)(packed + tile * 512))[lane];
}

// One full 3-layer MLP for a 16-row tile, executed by one wave.
// a0: layer-0 A fragment (K padded to 32). hA/hB: per-wave LDS tiles (16x64 f16).
// B fragments come from pre-packed f16 tiles: BT0 (4 tiles), BT1 (8 tiles), BT2 (2 tiles).
template<int BT0, int BT1, int BT2>
__device__ __forceinline__ v8f mlp3_wave(v16h a0, const _Float16* packed,
                                         const float* b0, const float* b1,
                                         _Float16* hA, _Float16* hB, int lane) {
    const int l16 = lane & 15;
    const int hi  = lane >> 4;
    v8f zero = {0.f,0.f,0.f,0.f,0.f,0.f,0.f,0.f};

    // ---- layer 0: [16 x Kin(pad32)] @ [32 x 64] -> 16x64 ----
    v8f h1[4];
#pragma unroll
    for (int t = 0; t < 4; ++t)
        h1[t] = wmma_f32_f16(a0, load_btile(packed, BT0 + t, lane), zero);
#pragma unroll
    for (int t = 0; t < 4; ++t) {
        const int   N    = t * 16 + l16;
        const float bias = b0[N];
#pragma unroll
        for (int r = 0; r < 8; ++r) {
            const int M = r + 8 * hi;                       // C layout: lane holds col N, rows r+8*hi
            hA[M * 64 + N] = (_Float16)gelu_exact(h1[t][r] + bias);
        }
    }
    // Re-stripe 16x64 hidden into interleaved A layout (K = (h&7) + 8*hi + 16*(h>>3))
    v16h aLo, aHi;
#pragma unroll
    for (int h = 0; h < 16; ++h) {
        const int k = (h & 7) + 8 * hi + 16 * (h >> 3);
        aLo[h] = hA[l16 * 64 + k];
        aHi[h] = hA[l16 * 64 + 32 + k];
    }

    // ---- layer 1: [16 x 64] @ [64 x 64] -> 16x64 ----
    v8f h2[4];
#pragma unroll
    for (int t = 0; t < 4; ++t)
        h2[t] = wmma_f32_f16(aHi, load_btile(packed, BT1 + t * 2 + 1, lane),
                wmma_f32_f16(aLo, load_btile(packed, BT1 + t * 2,     lane), zero));
#pragma unroll
    for (int t = 0; t < 4; ++t) {
        const int   N    = t * 16 + l16;
        const float bias = b1[N];
#pragma unroll
        for (int r = 0; r < 8; ++r) {
            const int M = r + 8 * hi;
            hB[M * 64 + N] = (_Float16)gelu_exact(h2[t][r] + bias);
        }
    }
    v16h cLo, cHi;
#pragma unroll
    for (int h = 0; h < 16; ++h) {
        const int k = (h & 7) + 8 * hi + 16 * (h >> 3);
        cLo[h] = hB[l16 * 64 + k];
        cHi[h] = hB[l16 * 64 + 32 + k];
    }

    // ---- layer 2: [16 x 64] @ [64 x 16(pad of 5)] -> 16x16 (cols 0..4 valid) ----
    return wmma_f32_f16(cHi, load_btile(packed, BT2 + 1, lane),
           wmma_f32_f16(cLo, load_btile(packed, BT2,     lane), zero));
}

// ---------------- weight packing: f32 [K][N] -> f16 WMMA B fragments ----------------
// B layout (16-bit, 16x16x32): lane holds column N = lane%16; half h holds K = (h&7) + 8*(lane/16) + 16*(h>>3).
__global__ void pack_weights_kernel(const float* phiW0, const float* phiW1, const float* phiW2,
                                    const float* psiW0, const float* psiW1, const float* psiW2,
                                    _Float16* packed) {
    const float* srcs[6] = {psiW0, psiW1, psiW2, phiW0, phiW1, phiW2};
    const int Ks[6]  = {6, 64, 64, 3, 64, 64};
    const int Ns[6]  = {64, 64, 5, 64, 64, 5};
    const int nN[6]  = {4, 4, 1, 4, 4, 1};
    const int nK[6]  = {1, 2, 2, 1, 2, 2};
    int tile = 0;
    for (int s = 0; s < 6; ++s) {
        for (int nt = 0; nt < nN[s]; ++nt)
            for (int kt = 0; kt < nK[s]; ++kt) {
                for (int e = threadIdx.x; e < 512; e += blockDim.x) {
                    const int lane = e >> 4, h = e & 15;
                    const int K = kt * 32 + (h & 7) + 8 * (lane >> 4) + 16 * (h >> 3);
                    const int N = nt * 16 + (lane & 15);
                    float v = (K < Ks[s] && N < Ns[s]) ? srcs[s][K * Ns[s] + N] : 0.0f;
                    packed[tile * 512 + e] = (_Float16)v;
                }
                ++tile;
            }
    }
}

// ---------------- phi stream: one wave per batch (16 particles = 1 tile) ----------------
__global__ void __launch_bounds__(256) phi_kernel(const float* __restrict__ x,
                                                  const float* __restrict__ b0,
                                                  const float* __restrict__ b1,
                                                  const float* __restrict__ b2,
                                                  const _Float16* __restrict__ packed,
                                                  float* __restrict__ phi_mean) {
    __shared__ _Float16 hA[8][16 * 64];
    __shared__ _Float16 hB[8][16 * 64];
    __shared__ float    red[8][2][16];
    const int wave = threadIdx.x >> 5, lane = threadIdx.x & 31;
    const int l16 = lane & 15, hi = lane >> 4;
    const int b = blockIdx.x * 8 + wave;

    const float* xb = x + (size_t)b * (NPART * 2);
    const float xs0 = xb[2 * l16 + 0];                  // sqrt(OMEGA)=1
    const float xs1 = xb[2 * l16 + 1];
    const float r2  = xs0 * xs0 + xs1 * xs1;

    v16h a0;
#pragma unroll
    for (int h = 0; h < 16; ++h) a0[h] = (_Float16)0.f;
    a0[0] = (hi == 0) ? (_Float16)xs0 : (_Float16)0.f;  // K=0..2 live only in lanes 0-15
    a0[1] = (hi == 0) ? (_Float16)xs1 : (_Float16)0.f;
    a0[2] = (hi == 0) ? (_Float16)r2  : (_Float16)0.f;

    v8f out = mlp3_wave<14, 18, 26>(a0, packed, b0, b1, hA[wave], hB[wave], lane);

    float colsum = 0.f;
#pragma unroll
    for (int r = 0; r < 8; ++r) colsum += out[r];       // all 16 rows valid
    red[wave][hi][l16] = colsum;
    __syncthreads();
    const float tot = red[wave][0][l16] + red[wave][1][l16];
    if (hi == 0 && l16 < DL)
        phi_mean[(size_t)b * DL + l16] = tot * (1.0f / 16.0f) + b2[l16];
}

// ---------------- psi stream: one block per batch, 8 waves = 8 pair tiles ----------------
__global__ void __launch_bounds__(256) psi_kernel(const float* __restrict__ x,
                                                  const float* __restrict__ b0,
                                                  const float* __restrict__ b1,
                                                  const float* __restrict__ b2,
                                                  const _Float16* __restrict__ packed,
                                                  float* __restrict__ psi_mean,
                                                  float* __restrict__ cusp_out) {
    __shared__ _Float16 hA[8][16 * 64];
    __shared__ _Float16 hB[8][16 * 64];
    __shared__ float    red[8];
    const int b = blockIdx.x;
    const int wave = threadIdx.x >> 5, lane = threadIdx.x & 31;
    const int l16 = lane & 15, hi = lane >> 4;
    if (threadIdx.x < 8) red[threadIdx.x] = 0.f;
    __syncthreads();

    const int p  = wave * 16 + l16;                     // 0..127 (120 real pairs)
    const int pc = (p < NPAIR) ? p : 0;
    int i = 0, rem = pc, cnt = NPART - 1;               // triu_indices(16,1) row-major
    while (rem >= cnt) { rem -= cnt; --cnt; ++i; }
    const int j = i + 1 + rem;

    const float* xb = x + (size_t)b * (NPART * 2);
    const float dx = xb[2 * i + 0] - xb[2 * j + 0];
    const float dy = xb[2 * i + 1] - xb[2 * j + 1];
    const float r  = __builtin_amdgcn_sqrtf(fmaf(dx, dx, fmaf(dy, dy, 1e-12f)));

    const float er     = __builtin_amdgcn_exp2f(-LOG2E * r);            // exp(-r)
    const float inv1pr = __builtin_amdgcn_rcpf(1.0f + r);
    float f[6];
    f[0] = __builtin_amdgcn_logf(1.0f + r) * LN2;                       // log1p(r)
    f[1] = r * inv1pr;                                                  // r/(1+r)
    f[2] = __builtin_amdgcn_exp2f(-LOG2E * r * r);                      // exp(-r^2)
    f[3] = __builtin_amdgcn_exp2f(-0.5f * LOG2E * r);                   // exp(-r/2)
    f[4] = er;                                                          // exp(-r)
    f[5] = er * er;                                                     // exp(-2r)

    v16h a0;
#pragma unroll
    for (int h = 0; h < 16; ++h) a0[h] = (_Float16)0.f;
#pragma unroll
    for (int c = 0; c < 6; ++c)
        a0[c] = (hi == 0) ? (_Float16)f[c] : (_Float16)0.f;   // K=0..5 live only in lanes 0-15

    v8f out = mlp3_wave<0, 4, 12>(a0, packed, b0, b1, hA[wave], hB[wave], lane);

    // row-masked (pad pairs >= 120) column sums; lane holds column n = l16
    float colsum = 0.f;
#pragma unroll
    for (int r8 = 0; r8 < 8; ++r8) {
        const int pm = wave * 16 + (r8 + 8 * hi);
        colsum += (pm < NPAIR) ? out[r8] : 0.f;
    }
    if (l16 < DL) atomicAdd(&red[l16], colsum);
    if (hi == 0 && p < NPAIR) atomicAdd(&red[5], r * er);     // cusp term r*exp(-r)
    __syncthreads();
    if (threadIdx.x < DL)
        psi_mean[(size_t)b * DL + threadIdx.x] =
            red[threadIdx.x] * (1.0f / (float)NPAIR) + b2[threadIdx.x];
    if (threadIdx.x == DL)
        cusp_out[b] = CUSP_GAMMA * red[5];
}

// ---------------- readout: one thread per batch (tiny) ----------------
__global__ void __launch_bounds__(256) rho_kernel(const float* __restrict__ d_emb,
                                                  const float* __restrict__ W0,
                                                  const float* __restrict__ rb0,
                                                  const float* __restrict__ W1,
                                                  const float* __restrict__ rb1,
                                                  const float* __restrict__ phi_mean,
                                                  const float* __restrict__ psi_mean,
                                                  const float* __restrict__ cusp,
                                                  float* __restrict__ out) {
    __shared__ float W0s[26 * 64];
    __shared__ float b0s[64], W1s[64];
    for (int e = threadIdx.x; e < 26 * 64; e += 256) W0s[e] = W0[e];
    if (threadIdx.x < 64) { b0s[threadIdx.x] = rb0[threadIdx.x]; W1s[threadIdx.x] = W1[threadIdx.x]; }
    __syncthreads();

    const int b = blockIdx.x * 256 + threadIdx.x;
    float in[26];
#pragma unroll
    for (int k = 0; k < 5; ++k)  in[k]      = phi_mean[(size_t)b * DL + k];
#pragma unroll
    for (int k = 0; k < 5; ++k)  in[5 + k]  = psi_mean[(size_t)b * DL + k];
#pragma unroll
    for (int k = 0; k < 16; ++k) in[10 + k] = d_emb[(size_t)b * DEMB + k];

    float acc = rb1[0];
    for (int h = 0; h < 64; ++h) {
        float s = b0s[h];
#pragma unroll
        for (int k = 0; k < 26; ++k) s = fmaf(in[k], W0s[k * 64 + h], s);
        acc = fmaf(gelu_exact(s), W1s[h], acc);
    }
    out[b] = acc + cusp[b];
}

extern "C" void kernel_launch(void* const* d_in, const int* in_sizes, int n_in,
                              void* d_out, int out_size, void* d_ws, size_t ws_size,
                              hipStream_t stream) {
    const float* x     = (const float*)d_in[0];
    const float* d_emb = (const float*)d_in[1];
    const float* phiW0 = (const float*)d_in[2];  const float* phib0 = (const float*)d_in[3];
    const float* phiW1 = (const float*)d_in[4];  const float* phib1 = (const float*)d_in[5];
    const float* phiW2 = (const float*)d_in[6];  const float* phib2 = (const float*)d_in[7];
    const float* psiW0 = (const float*)d_in[8];  const float* psib0 = (const float*)d_in[9];
    const float* psiW1 = (const float*)d_in[10]; const float* psib1 = (const float*)d_in[11];
    const float* psiW2 = (const float*)d_in[12]; const float* psib2 = (const float*)d_in[13];
    const float* rhoW0 = (const float*)d_in[14]; const float* rhob0 = (const float*)d_in[15];
    const float* rhoW1 = (const float*)d_in[16]; const float* rhob1 = (const float*)d_in[17];
    float* out = (float*)d_out;

    // workspace layout: [0,28672) packed f16 B tiles (pad to 32KB), then per-batch scratch
    _Float16* packed  = (_Float16*)d_ws;
    float* phi_mean   = (float*)((char*)d_ws + 32768);
    float* psi_mean   = phi_mean + (size_t)BATCH * DL;
    float* cusp       = psi_mean + (size_t)BATCH * DL;

    pack_weights_kernel<<<1, 256, 0, stream>>>(phiW0, phiW1, phiW2, psiW0, psiW1, psiW2, packed);
    phi_kernel<<<BATCH / 8, 256, 0, stream>>>(x, phib0, phib1, phib2, packed, phi_mean);
    psi_kernel<<<BATCH,     256, 0, stream>>>(x, psib0, psib1, psib2, packed, psi_mean, cusp);
    rho_kernel<<<BATCH/256, 256, 0, stream>>>(d_emb, rhoW0, rhob0, rhoW1, rhob1,
                                              phi_mean, psi_mean, cusp, out);
}
